// MLA_79886391705647
// MI455X (gfx1250) — matrix operand order repaired and checked by reading, not compiled, not run on hardware
//
#include <hip/hip_runtime.h>

// ---------------- problem constants ----------------
#define S      2048
#define H      2048
#define NH     16
#define NKV    8
#define Q_LORA 1536
#define KV_LORA 512
#define ROPE_D 64
#define NROPE_D 128
#define V_D    128
#define Q_HD   192          // NROPE_D + ROPE_D
#define KV_HD  256          // NROPE_D + V_D

typedef __attribute__((ext_vector_type(16))) __bf16 v16bf;
typedef __attribute__((ext_vector_type(8)))  float  v8f;
typedef __attribute__((ext_vector_type(4)))  unsigned v4u;
typedef __attribute__((ext_vector_type(8)))  int    v8i;
typedef __attribute__((ext_vector_type(4)))  int    v4i;

#if defined(__has_builtin)
#if __has_builtin(__builtin_amdgcn_tensor_load_to_lds) && \
    __has_builtin(__builtin_amdgcn_s_wait_tensorcnt)
#define USE_TDM 1
#endif
#endif

static __device__ inline unsigned short f2bf(float f) {
  union { float f; unsigned u; } v; v.f = f;
  unsigned r = v.u + 0x7FFFu + ((v.u >> 16) & 1u);
  return (unsigned short)(r >> 16);
}

// ---------------- f32 -> bf16 convert ----------------
__global__ __launch_bounds__(256) void k_f32_to_bf16(const float* __restrict__ x,
                                                     unsigned short* __restrict__ y,
                                                     long n) {
  long i = (long)blockIdx.x * 256 + threadIdx.x;
  long stride = (long)gridDim.x * 256;
  for (; i < n; i += stride) y[i] = f2bf(x[i]);
}

// ---------------- RMSNorm (f32 in, bf16 out) ----------------
__global__ __launch_bounds__(256) void k_rmsnorm(const float* __restrict__ x,
                                                 const float* __restrict__ w,
                                                 unsigned short* __restrict__ y,
                                                 int D, int ldx, int ldy) {
  int row = blockIdx.x;
  const float* xr = x + (long)row * ldx;
  __shared__ float red[256];
  float s = 0.f;
  for (int d = threadIdx.x; d < D; d += 256) { float v = xr[d]; s += v * v; }
  red[threadIdx.x] = s; __syncthreads();
  for (int off = 128; off; off >>= 1) {
    if (threadIdx.x < off) red[threadIdx.x] += red[threadIdx.x + off];
    __syncthreads();
  }
  float inv = rsqrtf(red[0] / (float)D + 1e-6f);
  unsigned short* yr = y + (long)row * ldy;
  for (int d = threadIdx.x; d < D; d += 256) yr[d] = f2bf(xr[d] * inv * w[d]);
}

// ---------------- generic bf16 WMMA GEMM ----------------
// C[z] = alpha * A[z] @ B[z / bDiv]
// A: row-major MxK (bf16, or f32 converted on the fly when A_F32)
// B: row-major KxN bf16 ; C: row-major MxN (f32 or bf16)
#define TM 128
#define TN 64
#define TK 32

template <bool A_F32, bool OUT_BF16>
__global__ __launch_bounds__(256) void k_wmma_gemm(
    const void* __restrict__ Av, const unsigned short* __restrict__ B,
    void* __restrict__ Cv, int M, int N, int K, int lda, int ldb, int ldc,
    float alpha, long aStride, long bStride, long cStride, int bDiv) {
  __shared__ __align__(16) unsigned short lsA[TM * TK];   // [128][32] row-major
  __shared__ __align__(16) unsigned short lsBT[TN * TK];  // [64][32]  (n, k) transposed

  const int tid  = threadIdx.x;
  const int lane = tid & 31;
  const int w    = tid >> 5;          // 0..7
  const int wm0  = (w >> 1) * 32;     // wave row offset in block tile
  const int wn0  = (w & 1) * 32;      // wave col offset in block tile

  const int bm = blockIdx.y * TM;
  const int bn = blockIdx.x * TN;
  const int z  = blockIdx.z;

  const unsigned short* Bb = B + (long)(z / bDiv) * bStride;

  v8f zero = {0.f, 0.f, 0.f, 0.f, 0.f, 0.f, 0.f, 0.f};
  v8f acc[2][2];
  acc[0][0] = zero; acc[0][1] = zero; acc[1][0] = zero; acc[1][1] = zero;

  const int arow = tid >> 1;        // 0..127
  const int acol = (tid & 1) * 16;  // 0 or 16
  const int bk   = tid >> 3;        // 0..31
  const int bn8  = (tid & 7) * 8;   // 0..56

  for (int k0 = 0; k0 < K; k0 += TK) {
    // ---- stage A tile [128 x 32] into LDS (row-major) ----
    if constexpr (A_F32) {
      const float* Af = ((const float*)Av) + (long)z * aStride;
      const float* src = Af + (long)(bm + arow) * lda + k0 + acol;
      float4 f0 = ((const float4*)src)[0];
      float4 f1 = ((const float4*)src)[1];
      float4 f2 = ((const float4*)src)[2];
      float4 f3 = ((const float4*)src)[3];
      union { uint4 u[2]; unsigned short s[16]; } t;
      t.s[0]=f2bf(f0.x); t.s[1]=f2bf(f0.y); t.s[2]=f2bf(f0.z); t.s[3]=f2bf(f0.w);
      t.s[4]=f2bf(f1.x); t.s[5]=f2bf(f1.y); t.s[6]=f2bf(f1.z); t.s[7]=f2bf(f1.w);
      t.s[8]=f2bf(f2.x); t.s[9]=f2bf(f2.y); t.s[10]=f2bf(f2.z); t.s[11]=f2bf(f2.w);
      t.s[12]=f2bf(f3.x); t.s[13]=f2bf(f3.y); t.s[14]=f2bf(f3.z); t.s[15]=f2bf(f3.w);
      *(uint4*)&lsA[arow * TK + acol]     = t.u[0];
      *(uint4*)&lsA[arow * TK + acol + 8] = t.u[1];
    } else {
#ifdef USE_TDM
      // Tensor Data Mover: wave 0 issues one 2D tile descriptor (128x32 bf16,
      // row stride lda) -> LDS, then drains TENSORcnt before the barrier.
      // Wave-uniform branch: waves 1..7 skip via s_cbranch_execz (TDM ignores
      // EXEC, so it must be branched around, not lane-masked).
      if (w == 0) {
        const unsigned short* Ab = ((const unsigned short*)Av) + (long)z * aStride;
        unsigned long long ga =
            (unsigned long long)(const void*)(Ab + (long)bm * lda + k0);
        v4u g0;
        g0.x = 1u;                               // count=1, no gather/restore
        g0.y = (unsigned)(size_t)&lsA[0];        // LDS byte address
        g0.z = (unsigned)ga;                     // global_addr[31:0]
        g0.w = (unsigned)((ga >> 32) & 0x01FFFFFFULL) | (2u << 30); // [56:32]|type=2
        v8i g1;
        g1[0] = 0x10000;           // data_size=1 (2 bytes), mask/flags=0
        g1[1] = (int)(32u << 16);  // tensor_dim0 = 32 (exact tile -> no OOB)
        g1[2] = (int)(128u << 16); // tensor_dim0 hi=0 | tensor_dim1 = 128
        g1[3] = (int)(32u << 16);  // tensor_dim1 hi=0 | tile_dim0 = 32
        g1[4] = 128;               // tile_dim1 = 128, tile_dim2 = 0
        g1[5] = lda;               // tensor_dim0_stride (elements)
        g1[6] = 0; g1[7] = 0;
        v4i gz4 = {0, 0, 0, 0};
        v8i gz8 = {0, 0, 0, 0, 0, 0, 0, 0};
        __builtin_amdgcn_tensor_load_to_lds(g0, g1, gz4, gz4, gz8, 0);
        __builtin_amdgcn_s_wait_tensorcnt(0);
      }
#else
      const unsigned short* Ab = ((const unsigned short*)Av) + (long)z * aStride;
      const unsigned short* src = Ab + (long)(bm + arow) * lda + k0 + acol;
      *(uint4*)&lsA[arow * TK + acol]     = ((const uint4*)src)[0];
      *(uint4*)&lsA[arow * TK + acol + 8] = ((const uint4*)src)[1];
#endif
    }
    // ---- stage B tile [32 x 64] into LDS transposed -> (n, k) ----
    {
      const unsigned short* src = Bb + (long)(k0 + bk) * ldb + bn + bn8;
      union { uint4 v; unsigned short s[8]; } ub; ub.v = *(const uint4*)src;
#pragma unroll
      for (int j = 0; j < 8; ++j) lsBT[(bn8 + j) * TK + bk] = ub.s[j];
      if (k0 + TK < K) __builtin_prefetch(src + (long)TK * ldb, 0, 1);
    }
    __syncthreads();

    // ---- build fragments per ISA VGPR layouts ----
    const int fm = lane & 15;
    const int ka = (lane >> 4) * 8;    // A: lanes<16 -> K 0..7 / 16..23 ; lanes>=16 -> +8
    const int kb = (lane >> 4) * 8;    // B: lanes<16 -> K 0..7 / 16..23 ; lanes>=16 -> +8

    v16bf afrag[2], bfrag[2];
#pragma unroll
    for (int i = 0; i < 2; ++i) {
      const unsigned short* pa = &lsA[(wm0 + i * 16 + fm) * TK + ka];
      union { uint4 u[2]; v16bf v; } ua;
      ua.u[0] = *(const uint4*)pa;
      ua.u[1] = *(const uint4*)(pa + 16);
      afrag[i] = ua.v;
      const unsigned short* pb = &lsBT[(wn0 + i * 16 + fm) * TK + kb];
      union { uint4 u[2]; v16bf v; } ub;
      ub.u[0] = *(const uint4*)pb;
      ub.u[1] = *(const uint4*)(pb + 16);
      bfrag[i] = ub.v;
    }
#pragma unroll
    for (int i = 0; i < 2; ++i)
#pragma unroll
      for (int j = 0; j < 2; ++j)
        acc[i][j] = __builtin_amdgcn_wmma_f32_16x16x32_bf16(
            false, afrag[i], false, bfrag[j], (short)0, acc[i][j], false, false);

    __syncthreads();
  }

  // ---- store C (f32 C/D layout: lanes<16 -> M=r, lanes>=16 -> M=r+8; N=lane&15) ----
  const int rbase = bm + wm0 + ((lane >> 4) * 8);
  const int cn    = bn + wn0 + (lane & 15);
  if constexpr (OUT_BF16) {
    unsigned short* C = ((unsigned short*)Cv) + (long)z * cStride;
#pragma unroll
    for (int i = 0; i < 2; ++i)
#pragma unroll
      for (int j = 0; j < 2; ++j)
#pragma unroll
        for (int r = 0; r < 8; ++r)
          C[(long)(rbase + i * 16 + r) * ldc + cn + j * 16] = f2bf(alpha * acc[i][j][r]);
  } else {
    float* C = ((float*)Cv) + (long)z * cStride;
#pragma unroll
    for (int i = 0; i < 2; ++i)
#pragma unroll
      for (int j = 0; j < 2; ++j)
#pragma unroll
        for (int r = 0; r < 8; ++r)
          C[(long)(rbase + i * 16 + r) * ldc + cn + j * 16] = alpha * acc[i][j][r];
  }
}

// ---------------- RoPE + assemble q_states / K^T / V (bf16) ----------------
// qf:  [S, NH*Q_HD] f32 ; kva: [S, KV_LORA+ROPE_D] f32 ; kvf: [S, NKV*KV_HD] f32
// qs:  [NH][S][Q_HD] bf16 ; kT: [NKV][Q_HD][S] bf16 ; vb: [NKV][S][V_D] bf16
__global__ __launch_bounds__(256) void k_assemble(
    const float* __restrict__ qf, const float* __restrict__ kva,
    const float* __restrict__ kvf, unsigned short* __restrict__ qs,
    unsigned short* __restrict__ kT, unsigned short* __restrict__ vb) {
  const int s = blockIdx.x;
  const float t = (float)s;
  __shared__ float sh_sin[ROPE_D], sh_cos[ROPE_D], kr[ROPE_D];
  if (threadIdx.x < ROPE_D) {
    int d = threadIdx.x;
    int i = d & 31;
    float invf = powf(10000.0f, -(2.0f * (float)i) / (float)ROPE_D);
    float fr = t * invf;
    sh_sin[d] = sinf(fr); sh_cos[d] = cosf(fr);
  }
  __syncthreads();

  const float* q_row = qf + (long)s * (NH * Q_HD);
  for (int idx = threadIdx.x; idx < NH * Q_HD; idx += 256) {
    int h = idx / Q_HD, d = idx % Q_HD;
    float val;
    if (d < NROPE_D) {
      val = q_row[h * Q_HD + d];
    } else {
      int dr = d - NROPE_D;
      float x  = q_row[h * Q_HD + d];
      float xr = (dr < 32) ? -q_row[h * Q_HD + NROPE_D + dr + 32]
                           :  q_row[h * Q_HD + NROPE_D + dr - 32];
      val = x * sh_cos[dr] + xr * sh_sin[dr];
    }
    qs[(long)h * ((long)S * Q_HD) + (long)s * Q_HD + d] = f2bf(val);
  }

  if (threadIdx.x < ROPE_D) {
    int dr = threadIdx.x;
    const float* krow = kva + (long)s * (KV_LORA + ROPE_D) + KV_LORA;
    float x  = krow[dr];
    float xr = (dr < 32) ? -krow[dr + 32] : krow[dr - 32];
    kr[dr] = x * sh_cos[dr] + xr * sh_sin[dr];
  }
  __syncthreads();

  const float* kvrow = kvf + (long)s * (NKV * KV_HD);
  for (int idx = threadIdx.x; idx < NKV * Q_HD; idx += 256) {
    int kh = idx / Q_HD, d = idx % Q_HD;
    float val = (d < NROPE_D) ? kvrow[kh * KV_HD + d] : kr[d - NROPE_D];
    kT[(long)kh * ((long)Q_HD * S) + (long)d * S + s] = f2bf(val);
  }
  for (int idx = threadIdx.x; idx < NKV * V_D; idx += 256) {
    int kh = idx >> 7, d = idx & 127;
    vb[(long)kh * ((long)S * V_D) + (long)s * V_D + d] = f2bf(kvrow[kh * KV_HD + NROPE_D + d]);
  }
}

// ---------------- in-place row softmax ----------------
__global__ __launch_bounds__(256) void k_softmax(float* __restrict__ attn, int ncols) {
  float* r = attn + (long)blockIdx.x * ncols;
  __shared__ float red[256];
  const int tid = threadIdx.x;
  float m = -3.0e38f;
  for (int c = tid; c < ncols; c += 256) m = fmaxf(m, r[c]);
  red[tid] = m; __syncthreads();
  for (int off = 128; off; off >>= 1) {
    if (tid < off) red[tid] = fmaxf(red[tid], red[tid + off]);
    __syncthreads();
  }
  float M = red[0]; __syncthreads();
  float s = 0.f;
  for (int c = tid; c < ncols; c += 256) { float e = __expf(r[c] - M); r[c] = e; s += e; }
  red[tid] = s; __syncthreads();
  for (int off = 128; off; off >>= 1) {
    if (tid < off) red[tid] += red[tid + off];
    __syncthreads();
  }
  float inv = 1.0f / red[0];
  for (int c = tid; c < ncols; c += 256) r[c] *= inv;
}

// ---------------- host-side orchestration ----------------
extern "C" void kernel_launch(void* const* d_in, const int* in_sizes, int n_in,
                              void* d_out, int out_size, void* d_ws, size_t ws_size,
                              hipStream_t stream) {
  (void)in_sizes; (void)n_in; (void)out_size; (void)ws_size;
  const float* hs     = (const float*)d_in[0];
  const float* q_a_w  = (const float*)d_in[2];
  const float* q_nw   = (const float*)d_in[3];
  const float* q_b_w  = (const float*)d_in[4];
  const float* kv_a_w = (const float*)d_in[5];
  const float* kv_nw  = (const float*)d_in[6];
  const float* kv_b_w = (const float*)d_in[7];
  const float* out_w  = (const float*)d_in[8];

  // ws partition (256B aligned)
  size_t off = 0;
  auto alloc = [&](size_t bytes) -> void* {
    void* p = (char*)d_ws + off;
    off += (bytes + 255) & ~(size_t)255;
    return p;
  };
  unsigned short* hsb  = (unsigned short*)alloc((size_t)S * H * 2);
  unsigned short* wqa  = (unsigned short*)alloc((size_t)H * Q_LORA * 2);
  unsigned short* wqb  = (unsigned short*)alloc((size_t)Q_LORA * NH * Q_HD * 2);
  unsigned short* wkva = (unsigned short*)alloc((size_t)H * (KV_LORA + ROPE_D) * 2);
  unsigned short* wkvb = (unsigned short*)alloc((size_t)KV_LORA * NKV * KV_HD * 2);
  unsigned short* wow  = (unsigned short*)alloc((size_t)NH * V_D * H * 2);
  float*          qa   = (float*)alloc((size_t)S * Q_LORA * 4);
  unsigned short* qan  = (unsigned short*)alloc((size_t)S * Q_LORA * 2);
  float*          qf   = (float*)alloc((size_t)S * NH * Q_HD * 4);
  float*          kva  = (float*)alloc((size_t)S * (KV_LORA + ROPE_D) * 4);
  unsigned short* kvn  = (unsigned short*)alloc((size_t)S * KV_LORA * 2);
  float*          kvf  = (float*)alloc((size_t)S * NKV * KV_HD * 4);
  unsigned short* qs   = (unsigned short*)alloc((size_t)NH * S * Q_HD * 2);
  unsigned short* kT   = (unsigned short*)alloc((size_t)NKV * Q_HD * S * 2);
  unsigned short* vb   = (unsigned short*)alloc((size_t)NKV * S * V_D * 2);
  unsigned short* ctx  = (unsigned short*)alloc((size_t)S * NH * V_D * 2);

  float* out_f  = (float*)d_out;                       // [S, H]
  float* attn_f = out_f + (long)S * H;                 // [NH, S, S]

  auto cvt = [&](const float* src, unsigned short* dst, long n) {
    int blocks = (int)((n + 255) / 256);
    if (blocks > 16384) blocks = 16384;
    k_f32_to_bf16<<<blocks, 256, 0, stream>>>(src, dst, n);
  };
  cvt(hs, hsb, (long)S * H);
  cvt(q_a_w, wqa, (long)H * Q_LORA);
  cvt(q_b_w, wqb, (long)Q_LORA * NH * Q_HD);
  cvt(kv_a_w, wkva, (long)H * (KV_LORA + ROPE_D));
  cvt(kv_b_w, wkvb, (long)KV_LORA * NKV * KV_HD);
  cvt(out_w, wow, (long)NH * V_D * H);

  // GEMM1: qa = hs @ q_a_w   (2048 x 1536 x 2048)
  k_wmma_gemm<false, false><<<dim3(Q_LORA / TN, S / TM, 1), 256, 0, stream>>>(
      hsb, wqa, qa, S, Q_LORA, H, H, Q_LORA, Q_LORA, 1.0f, 0L, 0L, 0L, 1);
  // RMSNorm(q)
  k_rmsnorm<<<S, 256, 0, stream>>>(qa, q_nw, qan, Q_LORA, Q_LORA, Q_LORA);
  // GEMM2: qf = qan @ q_b_w  (2048 x 3072 x 1536)
  k_wmma_gemm<false, false><<<dim3((NH * Q_HD) / TN, S / TM, 1), 256, 0, stream>>>(
      qan, wqb, qf, S, NH * Q_HD, Q_LORA, Q_LORA, NH * Q_HD, NH * Q_HD, 1.0f, 0L, 0L, 0L, 1);
  // GEMM3: kva = hs @ kv_a_w (2048 x 576 x 2048)
  k_wmma_gemm<false, false><<<dim3((KV_LORA + ROPE_D) / TN, S / TM, 1), 256, 0, stream>>>(
      hsb, wkva, kva, S, KV_LORA + ROPE_D, H, H, KV_LORA + ROPE_D, KV_LORA + ROPE_D,
      1.0f, 0L, 0L, 0L, 1);
  // RMSNorm(kv compressed part)
  k_rmsnorm<<<S, 256, 0, stream>>>(kva, kv_nw, kvn, KV_LORA, KV_LORA + ROPE_D, KV_LORA);
  // GEMM4: kvf = kvn @ kv_b_w (2048 x 2048 x 512)
  k_wmma_gemm<false, false><<<dim3((NKV * KV_HD) / TN, S / TM, 1), 256, 0, stream>>>(
      kvn, wkvb, kvf, S, NKV * KV_HD, KV_LORA, KV_LORA, NKV * KV_HD, NKV * KV_HD,
      1.0f, 0L, 0L, 0L, 1);
  // RoPE + build q_states / K^T / V
  k_assemble<<<S, 256, 0, stream>>>(qf, kva, kvf, qs, kT, vb);

  // scores: attn[h] = (1/sqrt(192)) * qs[h] @ kT[h/2]   (16 x [2048 x 2048 x 192])
  const float scale = 0.07216878364870322f;  // 1/sqrt(192)
  k_wmma_gemm<false, false><<<dim3(S / TN, S / TM, NH), 256, 0, stream>>>(
      qs, kT, attn_f, S, S, Q_HD, Q_HD, S, S, scale,
      (long)S * Q_HD, (long)Q_HD * S, (long)S * S, 2);
  // softmax in place
  k_softmax<<<NH * S, 256, 0, stream>>>(attn_f, S);
  // ctx[:, h*128:(h+1)*128] = attn[h] @ vb[h/2]   (16 x [2048 x 128 x 2048]), A is f32
  k_wmma_gemm<true, true><<<dim3(V_D / TN, S / TM, NH), 256, 0, stream>>>(
      attn_f, vb, ctx, S, V_D, S, S, V_D, NH * V_D, 1.0f,
      (long)S * S, (long)S * V_D, (long)V_D, 2);
  // out = ctx @ out_w (2048 x 2048 x 2048)
  k_wmma_gemm<false, false><<<dim3(H / TN, S / TM, 1), 256, 0, stream>>>(
      ctx, wow, out_f, S, H, NH * V_D, NH * V_D, H, H, 1.0f, 0L, 0L, 0L, 1);
}
